// GraphConv_76235669504319
// MI455X (gfx1250) — compile-verified, matrix operand order checked by reading, not run on hardware
//
#include <hip/hip_runtime.h>
#include <math.h>

// ---------------- problem constants (from reference) ----------------
constexpr int N_NODES = 100000;              // == 6250 * 16, so row tiles are exact
constexpr int N_EDGES = 1600000;
constexpr int ETOT    = N_EDGES + N_NODES;   // edges + self loops
constexpr int VIS = 512, TXT = 768;
constexpr int INH = 64;                      // IN/2
constexpr int FEAT = 131;                    // 128 + 3
constexpr int FEAT_PAD = 160;                // padded to multiple of 32 for WMMA K-loop
constexpr int HID  = 128;
constexpr float NEG_SLOPE = 0.2f;

// ---------------- WMMA types ----------------
typedef __attribute__((ext_vector_type(16))) __bf16 v16bf;
typedef __attribute__((ext_vector_type(8)))  float  v8f;

__device__ inline float gelu_exact(float x) {
    return 0.5f * x * (1.f + erff(x * 0.70710678118654752f));
}

// two-sided float atomic max (int/uint ordering trick)
__device__ inline void atomicMaxF(float* addr, float val) {
    if (!(val < 0.f)) atomicMax((int*)addr, __builtin_bit_cast(int, val));
    else              atomicMin((unsigned int*)addr, __builtin_bit_cast(unsigned int, val));
}

// ---------------- weight packing: W (Kreal x Nc fp32) -> bf16 WMMA B fragments ----------
// P layout: frag = (colTile * nk + kstep) * 32 + lane ; 16 contiguous bf16 per frag.
// Element i of lane L in kstep j: K = j*32 + (i/8)*16 + (L/16)*8 + (i%8), col = colTile*16 + (L%16)
__global__ void pack_w(const float* __restrict__ W, int Kreal, int Kpad, int Nc,
                       __bf16* __restrict__ P) {
    int idx = blockIdx.x * blockDim.x + threadIdx.x;
    int nk = Kpad >> 5;
    int total = (Nc >> 4) * nk * 32 * 16;
    if (idx >= total) return;
    int i    = idx & 15;
    int lane = (idx >> 4) & 31;
    int j    = (idx >> 9) % nk;
    int t    = idx / (nk << 9);
    int k = (j << 5) + ((i >> 3) << 4) + ((lane >> 4) << 3) + (i & 7);
    int c = (t << 4) + (lane & 15);
    float v = (k < Kreal) ? W[(size_t)k * Nc + c] : 0.f;
    P[idx] = (__bf16)v;
}

// ---------------- WMMA GEMM: C = A(M x K fp32) @ W(K x Nc), W pre-packed bf16 -----------
// One wave computes a 16x64 C tile (4 accumulators). No bounds checks: M % 16 == 0,
// K % 32 == 0 (padded). A loaded as 4x float4 per K-step, B as contiguous v16bf frags.
__global__ void wmma_gemm(const float* __restrict__ A, int lda, int K,
                          const __bf16* __restrict__ P, int nColTiles,
                          const float* __restrict__ bias,
                          float* __restrict__ C, int ldc, int col0,
                          const int* __restrict__ row_map, int nRowTiles)
{
    const int wib  = threadIdx.x >> 5;
    const int lane = threadIdx.x & 31;
    const int colGroups = nColTiles >> 2;                 // 4 tiles per wave
    const int wid = blockIdx.x * (blockDim.x >> 5) + wib;
    const int rowTile = wid / colGroups;
    const int cg      = wid % colGroups;
    if (rowTile >= nRowTiles) return;                     // wave-uniform exit

    const int half = lane >> 4, l16 = lane & 15;
    const int nk = K >> 5;
    const float* Arow = A + (size_t)(rowTile * 16 + l16) * lda;
    const v16bf* Pv = (const v16bf*)P;

    v8f acc[4];
#pragma unroll
    for (int t = 0; t < 4; ++t) acc[t] = (v8f){0.f,0.f,0.f,0.f,0.f,0.f,0.f,0.f};

    for (int j = 0; j < nk; ++j) {
        const int kbase = (j << 5) + (half << 3);
        const float4 a0 = *(const float4*)(Arow + kbase);
        const float4 a1 = *(const float4*)(Arow + kbase + 4);
        const float4 a2 = *(const float4*)(Arow + kbase + 16);
        const float4 a3 = *(const float4*)(Arow + kbase + 20);
        v16bf va;
        va[0]=(__bf16)a0.x;  va[1]=(__bf16)a0.y;  va[2]=(__bf16)a0.z;  va[3]=(__bf16)a0.w;
        va[4]=(__bf16)a1.x;  va[5]=(__bf16)a1.y;  va[6]=(__bf16)a1.z;  va[7]=(__bf16)a1.w;
        va[8]=(__bf16)a2.x;  va[9]=(__bf16)a2.y;  va[10]=(__bf16)a2.z; va[11]=(__bf16)a2.w;
        va[12]=(__bf16)a3.x; va[13]=(__bf16)a3.y; va[14]=(__bf16)a3.z; va[15]=(__bf16)a3.w;

        v16bf vb[4];
#pragma unroll
        for (int t = 0; t < 4; ++t)
            vb[t] = Pv[((size_t)(cg * 4 + t) * nk + j) * 32 + lane];
#pragma unroll
        for (int t = 0; t < 4; ++t)
            acc[t] = __builtin_amdgcn_wmma_f32_16x16x32_bf16(
                         false, va, false, vb[t], (short)0, acc[t], false, false);
    }

#pragma unroll
    for (int t = 0; t < 4; ++t) {
        const int colW = (cg * 4 + t) * 16 + l16;
        const float bc = bias ? bias[colW] : 0.f;
#pragma unroll
        for (int r = 0; r < 8; ++r) {
            int row = rowTile * 16 + (half << 3) + r;     // C layout: VGPR r -> M = r + 8*half
            int orow = row_map ? row_map[row] : row;
            C[(size_t)orow * ldc + col0 + colW] = acc[t][r] + bc;
        }
    }
}

// ---------------- elementwise / scatter kernels ----------------
__global__ void fill_f(float* p, float v, int n) {
    int i = blockIdx.x * blockDim.x + threadIdx.x;
    if (i < n) p[i] = v;
}

__global__ void scatter_ars(const float* __restrict__ ars, const int* __restrict__ cidx,
                            float* __restrict__ node, int n) {
    int i = blockIdx.x * blockDim.x + threadIdx.x;
    if (i >= n) return;
    int r = cidx[i];
    size_t base = (size_t)r * FEAT_PAD + 128;
    node[base + 0] = ars[i * 3 + 0];
    node[base + 1] = ars[i * 3 + 1];
    node[base + 2] = ars[i * 3 + 2];
}

__global__ void gelu_all(float* p, int n) {
    int i = blockIdx.x * blockDim.x + threadIdx.x;
    if (i < n) p[i] = gelu_exact(p[i]);   // gelu(0)=0 keeps pad columns zero
}

template<int ACT>   // 1=gelu, 2=relu
__global__ void bias_act(const float* __restrict__ in, const float* __restrict__ bias,
                         float* __restrict__ out, int n) {
    int i = blockIdx.x * blockDim.x + threadIdx.x;
    if (i >= n) return;
    float v = in[i] + bias[i & (HID - 1)];
    if (ACT == 1) v = gelu_exact(v);
    else if (ACT == 2) v = fmaxf(v, 0.f);
    out[i] = v;
}

// ---------------- attention: per-node scores, one wave per node ----------------
__global__ void att_scores(const float* __restrict__ h, const float* __restrict__ as,
                           const float* __restrict__ ad,
                           float* __restrict__ a_s, float* __restrict__ a_d, int n) {
    int node = blockIdx.x * (blockDim.x >> 5) + (threadIdx.x >> 5);
    int lane = threadIdx.x & 31;
    if (node >= n) return;
    float ss = 0.f, sd = 0.f;
    for (int j = lane; j < HID; j += 32) {
        float v = h[(size_t)node * HID + j];
        ss += v * as[j];
        sd += v * ad[j];
    }
    for (int o = 16; o > 0; o >>= 1) {
        ss += __shfl_down(ss, o, 32);
        sd += __shfl_down(sd, o, 32);
    }
    if (lane == 0) { a_s[node] = ss; a_d[node] = sd; }
}

__device__ inline void edge_sd(const int* edges, int i, int E, int& s, int& d) {
    if (i < E) { s = edges[2 * i]; d = edges[2 * i + 1]; }
    else       { s = d = i - E; }
}

__global__ void edge_max(const int* __restrict__ edges, const float* __restrict__ a_s,
                         const float* __restrict__ a_d, float* __restrict__ emax,
                         int etot, int E) {
    int i = blockIdx.x * blockDim.x + threadIdx.x;
    if (i >= etot) return;
    int s, d; edge_sd(edges, i, E, s, d);
    float e = a_s[s] + a_d[d];
    e = (e > 0.f) ? e : e * NEG_SLOPE;
    atomicMaxF(&emax[d], e);
}

__global__ void edge_exp(const int* __restrict__ edges, const float* __restrict__ a_s,
                         const float* __restrict__ a_d, const float* __restrict__ emax,
                         float* __restrict__ denom, float* __restrict__ ee,
                         int etot, int E) {
    int i = blockIdx.x * blockDim.x + threadIdx.x;
    if (i >= etot) return;
    int s, d; edge_sd(edges, i, E, s, d);
    float e = a_s[s] + a_d[d];
    e = (e > 0.f) ? e : e * NEG_SLOPE;
    float x = expf(e - emax[d]);
    ee[i] = x;
    atomicAdd(&denom[d], x);
}

// one wave per edge: agg[d] += h[s] * alpha
__global__ void edge_agg(const int* __restrict__ edges, const float* __restrict__ ee,
                         const float* __restrict__ denom, const float* __restrict__ h,
                         float* __restrict__ agg, int etot, int E) {
    int i    = blockIdx.x * (blockDim.x >> 5) + (threadIdx.x >> 5);
    int lane = threadIdx.x & 31;
    if (i >= etot) return;
    int s, d; edge_sd(edges, i, E, s, d);
    float alpha = ee[i] / (denom[d] + 1e-16f);
    for (int j = lane; j < HID; j += 32)
        atomicAdd(&agg[(size_t)d * HID + j], h[(size_t)s * HID + j] * alpha);
}

// ---------------- heads + edge predictions ----------------
__global__ void heads(const float* __restrict__ out,
                      const float* __restrict__ Wa, const float* __restrict__ ba,
                      const float* __restrict__ Wd, const float* __restrict__ bd,
                      float* __restrict__ ang, float* __restrict__ dst, int n) {
    int node = blockIdx.x * blockDim.x + threadIdx.x;
    if (node >= n) return;
    float a0 = ba[0], a1 = ba[1], d0 = bd[0], d1 = bd[1];
    const float* row = out + (size_t)node * HID;
    for (int j = 0; j < HID; ++j) {
        float v = row[j];
        a0 += v * Wa[2 * j];     a1 += v * Wa[2 * j + 1];
        d0 += v * Wd[2 * j];     d1 += v * Wd[2 * j + 1];
    }
    ang[2 * node] = a0; ang[2 * node + 1] = a1;
    dst[2 * node] = d0; dst[2 * node + 1] = d1;
}

__global__ void edge_pred(const int* __restrict__ edges, const float* __restrict__ ang,
                          const float* __restrict__ dst, float* __restrict__ pred, int E) {
    int i = blockIdx.x * blockDim.x + threadIdx.x;
    if (i >= E) return;
    int s = edges[2 * i], d = edges[2 * i + 1];
    pred[2 * i]     = ang[2 * s] * ang[2 * d] + ang[2 * s + 1] * ang[2 * d + 1];
    pred[2 * i + 1] = dst[2 * s] * dst[2 * d] + dst[2 * s + 1] * dst[2 * d + 1];
}

// ---------------- launch ----------------
static inline int cdiv(int a, int b) { return (a + b - 1) / b; }

extern "C" void kernel_launch(void* const* d_in, const int* in_sizes, int n_in,
                              void* d_out, int out_size, void* d_ws, size_t ws_size,
                              hipStream_t stream) {
    const float* image = (const float*)d_in[0];
    const float* text  = (const float*)d_in[1];
    const int*   cidx  = (const int*)d_in[2];
    /* d_in[3] = gt_indices (unused) */
    const int*   edges = (const int*)d_in[4];
    const float* ars   = (const float*)d_in[5];
    const float* Wi = (const float*)d_in[6];   const float* bi = (const float*)d_in[7];
    const float* Wt = (const float*)d_in[8];   const float* bt = (const float*)d_in[9];
    const float* W0 = (const float*)d_in[10];  const float* as0 = (const float*)d_in[11];
    const float* ad0 = (const float*)d_in[12]; const float* b0 = (const float*)d_in[13];
    const float* W1 = (const float*)d_in[14];  const float* as1 = (const float*)d_in[15];
    const float* ad1 = (const float*)d_in[16]; const float* b1 = (const float*)d_in[17];
    const float* Wa = (const float*)d_in[18];  const float* ba = (const float*)d_in[19];
    const float* Wd = (const float*)d_in[20];  const float* bd = (const float*)d_in[21];

    float* outp  = (float*)d_out;                      // N*HID
    float* predp = outp + (size_t)N_NODES * HID;       // E*2

    // -------- workspace carve-up --------
    // packed bf16 weights first (32B-aligned fragment loads)
    __bf16* Pi = (__bf16*)d_ws;                        // 512*64
    __bf16* Pt = Pi + (size_t)VIS * INH;               // 768*64
    __bf16* P0 = Pt + (size_t)TXT * INH;               // 160*128 (K padded 131->160)
    __bf16* P1 = P0 + (size_t)FEAT_PAD * HID;          // 128*128
    float*  fbase = (float*)(P1 + (size_t)HID * HID);

    float* node  = fbase;                              // N*FEAT_PAD (later reused as x2)
    float* hproj = node  + (size_t)N_NODES * FEAT_PAD; // N*HID
    float* agg   = hproj + (size_t)N_NODES * HID;      // N*HID
    float* a_s   = agg   + (size_t)N_NODES * HID;      // N
    float* a_d   = a_s + N_NODES;
    float* emax  = a_d + N_NODES;
    float* denom = emax + N_NODES;
    float* ee    = denom + N_NODES;                    // ETOT
    float* ang   = ee + ETOT;                          // 2N
    float* dst_  = ang + 2 * N_NODES;                  // 2N

    const int rowTiles = N_NODES / 16;                 // 6250 exact
    const int T = 256;
    const int GT = 128;                                // 4 waves per GEMM block

    // -------- pack weights to bf16 WMMA fragments --------
    pack_w<<<cdiv(VIS * INH, T), T, 0, stream>>>(Wi, VIS, VIS, INH, Pi);
    pack_w<<<cdiv(TXT * INH, T), T, 0, stream>>>(Wt, TXT, TXT, INH, Pt);
    pack_w<<<cdiv(FEAT_PAD * HID, T), T, 0, stream>>>(W0, FEAT, FEAT_PAD, HID, P0);
    pack_w<<<cdiv(HID * HID, T), T, 0, stream>>>(W1, HID, HID, HID, P1);

    // -------- input projections -> node (N x FEAT_PAD, zero-padded) --------
    fill_f<<<cdiv(N_NODES * FEAT_PAD, T), T, 0, stream>>>(node, 0.f, N_NODES * FEAT_PAD);
    wmma_gemm<<<cdiv(rowTiles * 1, 4), GT, 0, stream>>>(image, VIS, VIS, Pi, INH / 16, bi,
                                                        node, FEAT_PAD, 0,   cidx, rowTiles);
    wmma_gemm<<<cdiv(rowTiles * 1, 4), GT, 0, stream>>>(text,  TXT, TXT, Pt, INH / 16, bt,
                                                        node, FEAT_PAD, INH, cidx, rowTiles);
    scatter_ars<<<cdiv(N_NODES, T), T, 0, stream>>>(ars, cidx, node, N_NODES);
    gelu_all<<<cdiv(N_NODES * FEAT_PAD, T), T, 0, stream>>>(node, N_NODES * FEAT_PAD);

    // -------- GAT layer 1 --------
    wmma_gemm<<<cdiv(rowTiles * 2, 4), GT, 0, stream>>>(node, FEAT_PAD, FEAT_PAD, P0, HID / 16,
                                                        nullptr, hproj, HID, 0, nullptr, rowTiles);
    att_scores<<<cdiv(N_NODES, 8), T, 0, stream>>>(hproj, as0, ad0, a_s, a_d, N_NODES);
    fill_f<<<cdiv(N_NODES, T), T, 0, stream>>>(emax, -3.0e38f, N_NODES);
    fill_f<<<cdiv(N_NODES, T), T, 0, stream>>>(denom, 0.f, N_NODES);
    fill_f<<<cdiv(N_NODES * HID, T), T, 0, stream>>>(agg, 0.f, N_NODES * HID);
    edge_max<<<cdiv(ETOT, T), T, 0, stream>>>(edges, a_s, a_d, emax, ETOT, N_EDGES);
    edge_exp<<<cdiv(ETOT, T), T, 0, stream>>>(edges, a_s, a_d, emax, denom, ee, ETOT, N_EDGES);
    edge_agg<<<cdiv(ETOT, 8), T, 0, stream>>>(edges, ee, denom, hproj, agg, ETOT, N_EDGES);
    bias_act<2><<<cdiv(N_NODES * HID, T), T, 0, stream>>>(agg, b0, node /*x2*/, N_NODES * HID);

    // -------- GAT layer 2 --------
    wmma_gemm<<<cdiv(rowTiles * 2, 4), GT, 0, stream>>>(node, HID, HID, P1, HID / 16,
                                                        nullptr, hproj, HID, 0, nullptr, rowTiles);
    att_scores<<<cdiv(N_NODES, 8), T, 0, stream>>>(hproj, as1, ad1, a_s, a_d, N_NODES);
    fill_f<<<cdiv(N_NODES, T), T, 0, stream>>>(emax, -3.0e38f, N_NODES);
    fill_f<<<cdiv(N_NODES, T), T, 0, stream>>>(denom, 0.f, N_NODES);
    fill_f<<<cdiv(N_NODES * HID, T), T, 0, stream>>>(agg, 0.f, N_NODES * HID);
    edge_max<<<cdiv(ETOT, T), T, 0, stream>>>(edges, a_s, a_d, emax, ETOT, N_EDGES);
    edge_exp<<<cdiv(ETOT, T), T, 0, stream>>>(edges, a_s, a_d, emax, denom, ee, ETOT, N_EDGES);
    edge_agg<<<cdiv(ETOT, 8), T, 0, stream>>>(edges, ee, denom, hproj, agg, ETOT, N_EDGES);
    bias_act<1><<<cdiv(N_NODES * HID, T), T, 0, stream>>>(agg, b1, outp, N_NODES * HID);

    // -------- heads + per-edge predictions --------
    heads<<<cdiv(N_NODES, T), T, 0, stream>>>(outp, Wa, ba, Wd, bd, ang, dst_, N_NODES);
    edge_pred<<<cdiv(N_EDGES, T), T, 0, stream>>>(edges, ang, dst_, predp, N_EDGES);

    (void)in_sizes; (void)n_in; (void)out_size; (void)ws_size;
}
